// Grapher_3289944949106
// MI455X (gfx1250) — compile-verified, matrix operand order checked by reading, not run on hardware
//
#include <hip/hip_runtime.h>
#include <hip/hip_bf16.h>
#include <math.h>

// ---------------------------------------------------------------------------
// Grapher block (ViG) for MI455X / gfx1250, wave32 + WMMA bf16.
//   B=8, DIM=256, H=W=56, HW=3136, N=25088 positions, ~23 GFLOP of GEMM.
// Pipeline:
//   0. fold BN into conv weights (bf16) + bias (f32)
//   1. transpose-convert x  [B,C,HW]f32 -> xb [B,HW,C]bf16
//   2. GEMM1 (256x256): h = W1'·xb + b1'      -> h  f32 [B,C,HW]
//   3. MRConv4d: cat=[h, max(0, h-parity_min)] -> cat bf16 [B,HW,2C]
//   4. GEMM2 (512x512)+GELU                    -> g  bf16 [B,HW,2C]
//   5. GEMM3 (256x512)+bias+residual           -> out f32 [B,C,HW]
// GEMMs: 128(M)x256(N)x32(K) LDS double-buffered tiles, 8 wave32/block,
//        64x64 tile per wave = 16 v_wmma_f32_16x16x32_bf16 per K-step.
// Tile staging uses GLOBAL_LOAD_ASYNC_TO_LDS_B128 (ASYNCcnt) when the
// toolchain exposes the builtin, else global->VGPR->ds_store fallback.
// ---------------------------------------------------------------------------

typedef __attribute__((ext_vector_type(16))) __bf16        bf16x16;
typedef __attribute__((ext_vector_type(8)))  float         f32x8;
typedef __attribute__((ext_vector_type(4)))  unsigned int  u32x4;

#define BATCH 8
#define DIM   256
#define DIM2  512
#define HW    3136   // 56*56

#if defined(__HIP_DEVICE_COMPILE__) && \
    __has_builtin(__builtin_amdgcn_global_load_async_to_lds_b128) && \
    __has_builtin(__builtin_amdgcn_s_wait_asynccnt)
#define USE_ASYNC_LDS 1
// Parameter types per hipcc diagnostic: arg0 = AS1 (global) pointer to
// 16-byte int vector, arg1 = LDS pointer, then imm offset + cpol.
typedef __attribute__((__vector_size__(16))) int v4i_async;
typedef __attribute__((address_space(1))) v4i_async* gvp;
typedef __attribute__((address_space(3))) v4i_async* lvp;
#else
#define USE_ASYNC_LDS 0
#endif

__device__ __forceinline__ unsigned short f2bf(float f) {
  unsigned int u = __float_as_uint(f);
  unsigned int r = u + 0x7fffu + ((u >> 16) & 1u);   // round-to-nearest-even
  return (unsigned short)(r >> 16);
}

// ---------------------------------------------------------------------------
// 0. Fold BN (s = g*rsqrt(v+eps)) into conv weight + bias.
// ---------------------------------------------------------------------------
__global__ __launch_bounds__(256)
void fold_bn(const float* __restrict__ W, const float* __restrict__ bsrc,
             const float* __restrict__ g, const float* __restrict__ be,
             const float* __restrict__ m, const float* __restrict__ v,
             unsigned short* __restrict__ Wb, float* __restrict__ bOut, int Cin) {
  const int o = blockIdx.x;
  const float s = g[o] * rsqrtf(v[o] + 1e-5f);
  const float* wr = W + (size_t)o * Cin;
  unsigned short* wo = Wb + (size_t)o * Cin;
  for (int c = threadIdx.x; c < Cin; c += 256) wo[c] = f2bf(wr[c] * s);
  if (threadIdx.x == 0) bOut[o] = bsrc[o] * s + be[o] - m[o] * s;
}

// ---------------------------------------------------------------------------
// 1. x [B,C,HW] f32 -> xb [B,HW,C] bf16 via 32x32 LDS tile transpose
// ---------------------------------------------------------------------------
__global__ __launch_bounds__(256)
void x_transpose_bf16(const float* __restrict__ x, unsigned short* __restrict__ xb) {
  __shared__ float t[32][33];
  const int p0 = blockIdx.x * 32, c0 = blockIdx.y * 32, b = blockIdx.z;
  const int col = threadIdx.x & 31, row0 = threadIdx.x >> 5;
#pragma unroll
  for (int i = 0; i < 4; i++) {
    int r = row0 + i * 8;
    t[r][col] = x[((size_t)b * DIM + (c0 + r)) * HW + p0 + col];
  }
  __syncthreads();
#pragma unroll
  for (int i = 0; i < 4; i++) {
    int r = row0 + i * 8;
    xb[((size_t)b * HW + (p0 + r)) * DIM + c0 + col] = f2bf(t[col][r]);
  }
}

// ---------------------------------------------------------------------------
// GEMM: D[o,n] = sum_c A[o,c]*Act[b,n,c] + bias[o]  (+ epilogue)
//   A: [M,KD] bf16 (BN-folded weights); Act: [B,HW,KD] bf16 (n-major).
//   EPI 0: store f32 [B,M,HW];  EPI 1: GELU -> bf16 [B,HW,512];
//   EPI 2: +residual -> f32 [B,256,HW].
// Block tile 128x256x32; waves as 2(M)x4(N); wave tile 64x64 = 4x4 frags.
// ---------------------------------------------------------------------------
template <int KD, int EPI>
__global__ __launch_bounds__(256)
void gemm_bf16(const unsigned short* __restrict__ A,
               const unsigned short* __restrict__ Act,
               const float* __restrict__ bias,
               float* __restrict__ outF,
               unsigned short* __restrict__ outB,
               const float* __restrict__ resid,
               int Mtotal) {
  __shared__ alignas(16) unsigned short As[2][128][32];
  __shared__ alignas(16) unsigned short Bs[2][256][32];

  const int tid   = threadIdx.x;
  const int lane  = tid & 31;
  const int wave  = tid >> 5;
  const int wm    = wave >> 2;            // 0..1 : 64 M rows
  const int wn    = wave & 3;             // 0..3 : 64 N cols
  const int l16   = lane & 15;
  const int khalf = (lane >> 4) * 8;      // K chunk base per ISA 16-bit layout

  const int mBlk = blockIdx.y * 128;
  const int nBlk = blockIdx.x * 256;
  const int b    = blockIdx.z;

  const unsigned short* Abase = A + (size_t)mBlk * KD;
  const unsigned short* Bbase = Act + (size_t)b * HW * KD;

  f32x8 acc[4][4];
#pragma unroll
  for (int i = 0; i < 4; i++)
#pragma unroll
    for (int j = 0; j < 4; j++) acc[i][j] = f32x8{0.f, 0.f, 0.f, 0.f, 0.f, 0.f, 0.f, 0.f};

#if USE_ASYNC_LDS
  // Pre-zero B rows that fall outside HW (EXEC-masked async lanes write
  // nothing, so stale LDS must be cleared once; those rows are never loaded).
  {
    const u32x4 z = {0u, 0u, 0u, 0u};
    for (int c = tid; c < 1024; c += 256) {
      const int r = c >> 2, kc = (c & 3) * 8;
      if (nBlk + r >= HW) {
        *(u32x4*)&Bs[0][r][kc] = z;
        *(u32x4*)&Bs[1][r][kc] = z;
      }
    }
  }
#endif

  auto loadTile = [&](int s, int k0) {
    // A tile: 128x32 = 512 16B chunks; B tile: 256x32 = 1024 16B chunks.
#pragma unroll
    for (int c = tid, it = 0; it < 2; ++it, c += 256) {
      const int r = c >> 2, kc = (c & 3) * 8;
#if USE_ASYNC_LDS
      __builtin_amdgcn_global_load_async_to_lds_b128(
          (gvp)(Abase + (size_t)r * KD + k0 + kc),
          (lvp)&As[s][r][kc], 0, 0);
#else
      u32x4 av = *(const u32x4*)(Abase + (size_t)r * KD + k0 + kc);
      *(u32x4*)&As[s][r][kc] = av;
#endif
    }
#pragma unroll
    for (int c = tid, it = 0; it < 4; ++it, c += 256) {
      const int r = c >> 2, kc = (c & 3) * 8;
      const int n = nBlk + r;
#if USE_ASYNC_LDS
      if (n < HW)
        __builtin_amdgcn_global_load_async_to_lds_b128(
            (gvp)(Bbase + (size_t)n * KD + k0 + kc),
            (lvp)&Bs[s][r][kc], 0, 0);
#else
      u32x4 bv = {0u, 0u, 0u, 0u};
      if (n < HW) bv = *(const u32x4*)(Bbase + (size_t)n * KD + k0 + kc);
      *(u32x4*)&Bs[s][r][kc] = bv;
#endif
    }
  };

  loadTile(0, 0);
#if USE_ASYNC_LDS
  __builtin_amdgcn_s_wait_asynccnt(0);
#endif
  __syncthreads();

  constexpr int KSTEPS = KD / 32;
  for (int ks = 0; ks < KSTEPS; ++ks) {
    const int s = ks & 1;
    if (ks + 1 < KSTEPS) {
      if (ks + 2 < KSTEPS) {  // global_prefetch_b8 for the tile after next
        __builtin_prefetch(Abase + (size_t)(tid >> 2) * KD + (ks + 2) * 32, 0, 1);
        __builtin_prefetch(Bbase + (size_t)(nBlk + (tid >> 2)) * KD + (ks + 2) * 32, 0, 1);
      }
      loadTile(s ^ 1, (ks + 1) * 32);
    }

    union Frag { bf16x16 v; u32x4 q[2]; } fa[4], fb[4];
#pragma unroll
    for (int i = 0; i < 4; i++) {
      const unsigned short* p = &As[s][wm * 64 + i * 16 + l16][0];
      fa[i].q[0] = *(const u32x4*)(p + khalf);
      fa[i].q[1] = *(const u32x4*)(p + khalf + 16);
    }
#pragma unroll
    for (int j = 0; j < 4; j++) {
      const unsigned short* p = &Bs[s][wn * 64 + j * 16 + l16][0];
      fb[j].q[0] = *(const u32x4*)(p + khalf);
      fb[j].q[1] = *(const u32x4*)(p + khalf + 16);
    }

#pragma unroll
    for (int i = 0; i < 4; i++)
#pragma unroll
      for (int j = 0; j < 4; j++)
        acc[i][j] = __builtin_amdgcn_wmma_f32_16x16x32_bf16(
            false, fa[i].v, false, fb[j].v, (short)0, acc[i][j], false, false);

#if USE_ASYNC_LDS
    __builtin_amdgcn_s_wait_asynccnt(0);
#endif
    __syncthreads();
  }

  // Epilogue. D frag: VGPR r, lanes0-15 -> M=r, lanes16-31 -> M=r+8; N=lane&15.
  const int oBase = mBlk + wm * 64 + (lane >> 4) * 8;
  const int nBase = nBlk + wn * 64 + l16;
#pragma unroll
  for (int i = 0; i < 4; i++) {
#pragma unroll
    for (int j = 0; j < 4; j++) {
      const int n = nBase + j * 16;
      if (n >= HW) continue;
#pragma unroll
      for (int r = 0; r < 8; r++) {
        const int o = oBase + i * 16 + r;
        float val = acc[i][j][r] + bias[o];
        if (EPI == 0) {
          outF[((size_t)b * Mtotal + o) * HW + n] = val;
        } else if (EPI == 1) {
          val = 0.5f * val * (1.0f + erff(val * 0.70710678118654752f));
          outB[((size_t)b * HW + n) * DIM2 + o] = f2bf(val);
        } else {
          const size_t idx = ((size_t)b * DIM + o) * HW + n;
          outF[idx] = val + resid[idx];
        }
      }
    }
  }
}

// ---------------------------------------------------------------------------
// 3. MRConv4d (K=2): xj = max(0, h - min over same-parity rows/cols != self)
// ---------------------------------------------------------------------------
__global__ __launch_bounds__(256)
void mrconv_cat(const float* __restrict__ h, unsigned short* __restrict__ cat) {
  __shared__ float tile[HW];
  const int c = blockIdx.x, b = blockIdx.y;
  const float* hp = h + ((size_t)b * DIM + c) * HW;
  for (int p = threadIdx.x; p < HW; p += 256) tile[p] = hp[p];
  __syncthreads();
  for (int p = threadIdx.x; p < HW; p += 256) {
    const int y = p / 56, x = p - y * 56;
    const float hv = tile[p];
    float mn = 3.4e38f;
#pragma unroll 4
    for (int i = 2; i < 56; i += 2) {
      int y2 = y + i; if (y2 >= 56) y2 -= 56;
      int x2 = x + i; if (x2 >= 56) x2 -= 56;
      mn = fminf(mn, tile[y2 * 56 + x]);
      mn = fminf(mn, tile[y * 56 + x2]);
    }
    const float xj = fmaxf(0.f, hv - mn);
    unsigned short* cp = cat + ((size_t)b * HW + p) * DIM2;
    cp[c]       = f2bf(hv);
    cp[c + DIM] = f2bf(xj);
  }
}

// ---------------------------------------------------------------------------
extern "C" void kernel_launch(void* const* d_in, const int* in_sizes, int n_in,
                              void* d_out, int out_size, void* d_ws, size_t ws_size,
                              hipStream_t stream) {
  const float* x   = (const float*)d_in[0];
  const float* w1  = (const float*)d_in[1];
  const float* b1  = (const float*)d_in[2];
  const float* g1  = (const float*)d_in[3];
  const float* be1 = (const float*)d_in[4];
  const float* m1  = (const float*)d_in[5];
  const float* v1  = (const float*)d_in[6];
  const float* wg  = (const float*)d_in[7];
  const float* bg  = (const float*)d_in[8];
  const float* gg  = (const float*)d_in[9];
  const float* beg = (const float*)d_in[10];
  const float* mg  = (const float*)d_in[11];
  const float* vg  = (const float*)d_in[12];
  const float* w2  = (const float*)d_in[13];
  const float* b2  = (const float*)d_in[14];
  const float* g2  = (const float*)d_in[15];
  const float* be2 = (const float*)d_in[16];
  const float* m2  = (const float*)d_in[17];
  const float* v2  = (const float*)d_in[18];

  char* ws = (char*)d_ws;
  size_t off = 0;
  auto alloc = [&](size_t bytes) { void* p = ws + off; off = (off + bytes + 255) & ~(size_t)255; return p; };
  unsigned short* w1b   = (unsigned short*)alloc((size_t)DIM  * DIM  * 2);
  unsigned short* wgb   = (unsigned short*)alloc((size_t)DIM2 * DIM2 * 2);
  unsigned short* w2b   = (unsigned short*)alloc((size_t)DIM  * DIM2 * 2);
  float*          bias1 = (float*)alloc(DIM  * 4);
  float*          biasg = (float*)alloc(DIM2 * 4);
  float*          bias2 = (float*)alloc(DIM  * 4);
  unsigned short* xb    = (unsigned short*)alloc((size_t)BATCH * HW * DIM  * 2);
  float*          hbuf  = (float*)alloc((size_t)BATCH * DIM * HW * 4);
  unsigned short* cat   = (unsigned short*)alloc((size_t)BATCH * HW * DIM2 * 2);
  unsigned short* gbuf  = (unsigned short*)alloc((size_t)BATCH * HW * DIM2 * 2);

  // 0. BN folding into weights/bias
  fold_bn<<<DIM,  256, 0, stream>>>(w1, b1, g1, be1, m1, v1, w1b, bias1, DIM);
  fold_bn<<<DIM2, 256, 0, stream>>>(wg, bg, gg, beg, mg, vg, wgb, biasg, DIM2);
  fold_bn<<<DIM,  256, 0, stream>>>(w2, b2, g2, be2, m2, v2, w2b, bias2, DIM2);

  // 1. x -> [B,HW,C] bf16
  x_transpose_bf16<<<dim3(HW / 32, DIM / 32, BATCH), 256, 0, stream>>>(x, xb);

  const int nTiles = (HW + 255) / 256;  // 13

  // 2. fc1
  gemm_bf16<DIM, 0><<<dim3(nTiles, DIM / 128, BATCH), 256, 0, stream>>>(
      w1b, xb, bias1, hbuf, nullptr, nullptr, DIM);

  // 3. MRConv4d + concat
  mrconv_cat<<<dim3(DIM, BATCH), 256, 0, stream>>>(hbuf, cat);

  // 4. grapher conv + GELU
  gemm_bf16<DIM2, 1><<<dim3(nTiles, DIM2 / 128, BATCH), 256, 0, stream>>>(
      wgb, cat, biasg, nullptr, gbuf, nullptr, DIM2);

  // 5. fc2 + residual
  gemm_bf16<DIM2, 2><<<dim3(nTiles, DIM / 128, BATCH), 256, 0, stream>>>(
      w2b, gbuf, bias2, (float*)d_out, nullptr, x, DIM);
}
